// LlamaAttentionForReg_24412594111193
// MI455X (gfx1250) — compile-verified, matrix operand order checked by reading, not run on hardware
//
#include <hip/hip_runtime.h>
#include <cstdint>
#include <cstddef>

// ---------------------------------------------------------------------------
// Problem constants (B=1)
// ---------------------------------------------------------------------------
#define S_LEN 2048
#define HDIM  4096
#define N_H   32
#define N_KV  8
#define HD_   128
#define GRP   4
#define RANK  16

typedef __bf16 bf16;
typedef __bf16 v16bf __attribute__((ext_vector_type(16)));
typedef float  v8f   __attribute__((ext_vector_type(8)));

static __device__ __forceinline__ v8f wmma_bf16(v16bf a, v16bf b, v8f c) {
  // D = A(16x32 bf16) * B(32x16 bf16) + C(16x16 f32)
  return __builtin_amdgcn_wmma_f32_16x16x32_bf16(false, a, false, b, (short)0, c,
                                                 false, false);
}

// wave32 xor-shuffle via ds_swizzle group-of-32 mode (and=0x1f, xor=XM).
template <int XM>
static __device__ __forceinline__ float swz_xor(float v) {
  int i = __builtin_amdgcn_ds_swizzle(__float_as_int(v), 0x1f | (XM << 10));
  return __int_as_float(i);
}

// ---------------------------------------------------------------------------
// fp32 -> bf16 cast (grid-stride)
// ---------------------------------------------------------------------------
__global__ void cast_f32_bf16(const float* __restrict__ in, bf16* __restrict__ out,
                              int n) {
  for (int i = blockIdx.x * blockDim.x + threadIdx.x; i < n;
       i += gridDim.x * blockDim.x)
    out[i] = (bf16)in[i];
}

// ---------------------------------------------------------------------------
// LoRA t = X @ a^T  ->  [M][16] fp32.   X fp32 [M][K]
// block = 64 threads: 4 rows x 16 ranks
// ---------------------------------------------------------------------------
__global__ void lora_t_f32(const float* __restrict__ X, const float* __restrict__ amat,
                           float* __restrict__ t, int K) {
  int m = blockIdx.x * 4 + (threadIdx.x >> 4);
  int r = threadIdx.x & 15;
  const float4* x = (const float4*)(X + (size_t)m * K);
  const float4* a = (const float4*)(amat + (size_t)r * K);
  float acc = 0.f;
  for (int k = 0; k < K / 4; ++k) {
    float4 xv = x[k], av = a[k];
    acc += xv.x * av.x + xv.y * av.y + xv.z * av.z + xv.w * av.w;
  }
  t[m * RANK + r] = acc;
}

// Same, X in bf16 (attention output feeding the O projection LoRA).
__global__ void lora_t_bf16(const bf16* __restrict__ X, const float* __restrict__ amat,
                            float* __restrict__ t, int K) {
  int m = blockIdx.x * 4 + (threadIdx.x >> 4);
  int r = threadIdx.x & 15;
  const bf16*  x = X + (size_t)m * K;
  const float* a = amat + (size_t)r * K;
  float acc = 0.f;
#pragma unroll 4
  for (int k = 0; k < K; ++k) acc += (float)x[k] * a[k];
  t[m * RANK + r] = acc;
}

// ---------------------------------------------------------------------------
// WMMA GEMM:  C[M][N] = A[M][K](bf16) @ W[N][K]^T(bf16) + t[M][16] @ bmat[N][16]^T
// Tile: 128(M) x 64(N) per WG, BK=32.  8 waves, each 32x32 (2x2 WMMA tiles).
// MODE 0: store fp32 row-major [M][N]
// MODE 1: store bf16 row-major [M][N]
// MODE 2: store bf16 transposed [N][ldT] (ldT = M)   (used for V^T)
// ---------------------------------------------------------------------------
template <int MODE>
__global__ __launch_bounds__(256) void gemm_lora(
    const bf16* __restrict__ A, const bf16* __restrict__ W,
    const float* __restrict__ tmat, const float* __restrict__ bmat,
    float* __restrict__ outF, bf16* __restrict__ outB, int N, int K, int ldT) {
  __shared__ bf16 As[128][32];
  __shared__ bf16 Bs[64][32];

  const int tid  = threadIdx.x;
  const int blkM = blockIdx.y, blkN = blockIdx.x;
  const int wave = tid >> 5, lane = tid & 31;
  const int lh = lane >> 4, lm = lane & 15;
  const int m0 = (wave & 3) * 32;   // wave's M offset in tile
  const int n0 = (wave >> 2) * 32;  // wave's N offset in tile

  // cooperative load mapping
  const int rA = tid >> 1, cA = (tid & 1) * 16;  // 128 rows x 32 cols, 32B/thread
  const int rB = tid >> 2, cB = (tid & 3) * 8;   // 64 rows x 32 cols, 16B/thread
  const bf16* Abase = A + (size_t)(blkM * 128 + rA) * K + cA;
  const bf16* Bbase = W + (size_t)(blkN * 64 + rB) * K + cB;

  v8f c00 = {}, c01 = {}, c10 = {}, c11 = {};

  for (int k0 = 0; k0 < K; k0 += 32) {
    uint4 a0 = *(const uint4*)(Abase + k0);
    uint4 a1 = *(const uint4*)(Abase + k0 + 8);
    uint4 b0 = *(const uint4*)(Bbase + k0);
    __syncthreads();  // protect previous iteration's LDS reads
    *(uint4*)&As[rA][cA]     = a0;
    *(uint4*)&As[rA][cA + 8] = a1;
    *(uint4*)&Bs[rB][cB]     = b0;
    __syncthreads();
    if (k0 + 32 < K) {
      __builtin_prefetch(Abase + k0 + 32, 0, 1);
      __builtin_prefetch(Bbase + k0 + 32, 0, 1);
    }
    // A-frag: row = m0+sub*16+lm, K-chunk = lh*16 (contiguous k = lh*16 + j, same order as B)
    v16bf alo = *(const v16bf*)&As[m0 + lm][lh * 16];
    v16bf ahi = *(const v16bf*)&As[m0 + 16 + lm][lh * 16];
    v16bf blo = *(const v16bf*)&Bs[n0 + lm][lh * 16];
    v16bf bhi = *(const v16bf*)&Bs[n0 + 16 + lm][lh * 16];
    c00 = wmma_bf16(alo, blo, c00);
    c01 = wmma_bf16(alo, bhi, c01);
    c10 = wmma_bf16(ahi, blo, c10);
    c11 = wmma_bf16(ahi, bhi, c11);
  }

  // Epilogue: D element e @ lane -> M = e + 8*lh, N = lm (per 16x16 tile).
  const int Mbase = blkM * 128 + m0 + 8 * lh;
  const int Nbase = blkN * 64 + n0 + lm;
#pragma unroll
  for (int sn = 0; sn < 2; ++sn) {
    const int Ng = Nbase + sn * 16;
    float bb[RANK];
#pragma unroll
    for (int r = 0; r < RANK; ++r) bb[r] = bmat[(size_t)Ng * RANK + r];
#pragma unroll
    for (int sm = 0; sm < 2; ++sm) {
      v8f cc = sm == 0 ? (sn == 0 ? c00 : c01) : (sn == 0 ? c10 : c11);
#pragma unroll
      for (int e = 0; e < 8; ++e) {
        const int Mg = Mbase + sm * 16 + e;
        float v = cc[e];
        const float* tr = tmat + (size_t)Mg * RANK;
#pragma unroll
        for (int r = 0; r < RANK; ++r) v += tr[r] * bb[r];
        if (MODE == 0)      outF[(size_t)Mg * N + Ng] = v;
        else if (MODE == 1) outB[(size_t)Mg * N + Ng] = (bf16)v;
        else                outB[(size_t)Ng * ldT + Mg] = (bf16)v;
      }
    }
  }
}

// ---------------------------------------------------------------------------
// In-place RoPE on bf16 buffer [S][nheads*128].
// ---------------------------------------------------------------------------
__global__ void rope_kernel(bf16* __restrict__ buf, const int* __restrict__ pos_ids,
                            int nheads, int ld) {
  int idx = blockIdx.x * blockDim.x + threadIdx.x;
  int total = S_LEN * nheads * 64;
  if (idx >= total) return;
  int d = idx & 63;
  int h = (idx >> 6) % nheads;
  int s = idx / (nheads * 64);
  float pos = (float)pos_ids[s];
  float inv = __expf(-(float)d * 0.14391157f);  // ln(10000)/64
  float ang = pos * inv;
  float c = __cosf(ang), si = __sinf(ang);
  bf16* row = buf + (size_t)s * ld + h * HD_;
  float x1 = (float)row[d], x2 = (float)row[d + 64];
  row[d]      = (bf16)(x1 * c - x2 * si);
  row[d + 64] = (bf16)(x2 * c + x1 * si);
}

// ---------------------------------------------------------------------------
// Flash attention.  Grid: (S/64, NH).  Block: 128 (4 independent waves).
// Wave w owns 16 query rows. Per 32-key step:
//   QK^T: 2 score tiles x 4 WMMAs; softmax via ds_swizzle row reductions;
//   P transposed through LDS (D-layout -> A-layout); PV: 8 WMMAs vs V^T.
// Causal mask applied analytically with -1e9 (== attention_mask contents).
// ---------------------------------------------------------------------------
__global__ __launch_bounds__(128) void attn_kernel(
    const bf16* __restrict__ Qb,   // [S][4096] bf16 (post-RoPE)
    const bf16* __restrict__ Kb,   // [S][1024] bf16 (post-RoPE)
    const bf16* __restrict__ Vt,   // [1024][S] bf16 (d-major)
    bf16* __restrict__ Ob) {       // [S][4096] bf16
  __shared__ bf16 pLds[4][16][32];  // per-wave P transpose buffer

  const int wave = threadIdx.x >> 5, lane = threadIdx.x & 31;
  const int lh = lane >> 4, lm = lane & 15;
  const int h   = blockIdx.y;
  const int kvh = h / GRP;
  const int q0  = blockIdx.x * 64 + wave * 16;
  const float scale = 0.08838834764831845f;  // 1/sqrt(128)

  // Q fragments: row q0+lm, d = c*32 + lh*16 + j
  v16bf qf[4];
  const bf16* qrow = Qb + (size_t)(q0 + lm) * HDIM + h * HD_ + lh * 16;
#pragma unroll
  for (int c = 0; c < 4; ++c) qf[c] = *(const v16bf*)(qrow + c * 32);

  float mrun[8], lrun[8];
  v8f acc[8];
#pragma unroll
  for (int e = 0; e < 8; ++e) { mrun[e] = -1e30f; lrun[e] = 0.f; }
#pragma unroll
  for (int dt = 0; dt < 8; ++dt) acc[dt] = (v8f){};

  const int nkb = (q0 + 16 + 31) >> 5;  // causal: only keys <= q0+15 needed
  for (int kb = 0; kb < nkb; ++kb) {
    const int kbase = kb * 32;
    // ---- scores: two 16x16 tiles (keys kbase..+15, +16..+31)
    v8f st0 = {}, st1 = {};
    const bf16* kr0 = Kb + (size_t)(kbase + lm) * (N_KV * HD_) + kvh * HD_ + lh * 16;
    const bf16* kr1 = kr0 + (size_t)16 * (N_KV * HD_);
#pragma unroll
    for (int c = 0; c < 4; ++c) {
      st0 = wmma_bf16(qf[c], *(const v16bf*)(kr0 + c * 32), st0);
      st1 = wmma_bf16(qf[c], *(const v16bf*)(kr1 + c * 32), st1);
    }
    // ---- streaming softmax (row = q0 + e + 8*lh, col = kbase(+16) + lm)
    float pr0[8], pr1[8], alpha[8];
#pragma unroll
    for (int e = 0; e < 8; ++e) {
      const int qidx = q0 + e + 8 * lh;
      float s0 = st0[e] * scale + ((kbase + lm)      <= qidx ? 0.f : -1e9f);
      float s1 = st1[e] * scale + ((kbase + 16 + lm) <= qidx ? 0.f : -1e9f);
      float rm = fmaxf(s0, s1);
      rm = fmaxf(rm, swz_xor<1>(rm));
      rm = fmaxf(rm, swz_xor<2>(rm));
      rm = fmaxf(rm, swz_xor<4>(rm));
      rm = fmaxf(rm, swz_xor<8>(rm));
      float mnew = fmaxf(mrun[e], rm);
      float a  = __expf(mrun[e] - mnew);
      float p0 = __expf(s0 - mnew);
      float p1 = __expf(s1 - mnew);
      float rs = p0 + p1;
      rs += swz_xor<1>(rs);
      rs += swz_xor<2>(rs);
      rs += swz_xor<4>(rs);
      rs += swz_xor<8>(rs);
      lrun[e] = lrun[e] * a + rs;
      mrun[e] = mnew;
      alpha[e] = a;
      pr0[e] = p0;
      pr1[e] = p1;
    }
#pragma unroll
    for (int dt = 0; dt < 8; ++dt)
#pragma unroll
      for (int e = 0; e < 8; ++e) acc[dt][e] *= alpha[e];
    // ---- P: D-layout -> A-layout via LDS (wave-private region)
#pragma unroll
    for (int e = 0; e < 8; ++e) {
      pLds[wave][e + 8 * lh][lm]      = (bf16)pr0[e];
      pLds[wave][e + 8 * lh][16 + lm] = (bf16)pr1[e];
    }
    asm volatile("s_wait_dscnt 0" ::: "memory");
    v16bf pf = *(const v16bf*)&pLds[wave][lm][lh * 16];
    // ---- PV: B-frag = V^T row (kvh*128 + dt*16 + lm), keys kbase + lh*16 + j
    const bf16* vr = Vt + (size_t)(kvh * HD_ + lm) * S_LEN + kbase + lh * 16;
#pragma unroll
    for (int dt = 0; dt < 8; ++dt)
      acc[dt] = wmma_bf16(pf, *(const v16bf*)(vr + (size_t)dt * 16 * S_LEN), acc[dt]);
  }

  // ---- normalize and store (row = q0 + e + 8*lh, col = h*128 + dt*16 + lm)
  float rinv[8];
#pragma unroll
  for (int e = 0; e < 8; ++e) rinv[e] = 1.f / lrun[e];
#pragma unroll
  for (int dt = 0; dt < 8; ++dt)
#pragma unroll
    for (int e = 0; e < 8; ++e)
      Ob[(size_t)(q0 + e + 8 * lh) * HDIM + h * HD_ + dt * 16 + lm] =
          (bf16)(acc[dt][e] * rinv[e]);
}

// ---------------------------------------------------------------------------
// Host-side orchestration
// ---------------------------------------------------------------------------
extern "C" void kernel_launch(void* const* d_in, const int* in_sizes, int n_in,
                              void* d_out, int out_size, void* d_ws, size_t ws_size,
                              hipStream_t stream) {
  const float* hidden = (const float*)d_in[0];
  const int*   posids = (const int*)d_in[2];
  const float* q_w = (const float*)d_in[3];
  const float* q_a = (const float*)d_in[4];
  const float* q_b = (const float*)d_in[5];
  const float* k_w = (const float*)d_in[6];
  const float* k_a = (const float*)d_in[7];
  const float* k_b = (const float*)d_in[8];
  const float* v_w = (const float*)d_in[9];
  const float* v_a = (const float*)d_in[10];
  const float* v_b = (const float*)d_in[11];
  const float* o_w = (const float*)d_in[12];
  const float* o_a = (const float*)d_in[13];
  const float* o_b = (const float*)d_in[14];
  float* outF = (float*)d_out;

  // workspace layout
  char* p = (char*)d_ws;
  size_t off = 0;
  auto take = [&](size_t bytes) {
    void* r = p + off;
    off = (off + bytes + 255) & ~(size_t)255;
    return r;
  };
  bf16*  Xb = (bf16*)take((size_t)S_LEN * HDIM * 2);
  bf16*  Wq = (bf16*)take((size_t)HDIM * HDIM * 2);
  bf16*  Wk = (bf16*)take((size_t)N_KV * HD_ * HDIM * 2);
  bf16*  Wv = (bf16*)take((size_t)N_KV * HD_ * HDIM * 2);
  bf16*  Wo = (bf16*)take((size_t)HDIM * HDIM * 2);
  bf16*  Qb = (bf16*)take((size_t)S_LEN * HDIM * 2);
  bf16*  Kb = (bf16*)take((size_t)S_LEN * N_KV * HD_ * 2);
  bf16*  Vt = (bf16*)take((size_t)N_KV * HD_ * S_LEN * 2);
  bf16*  AO = (bf16*)take((size_t)S_LEN * HDIM * 2);
  float* tq = (float*)take((size_t)S_LEN * RANK * 4);
  float* tk = (float*)take((size_t)S_LEN * RANK * 4);
  float* tv = (float*)take((size_t)S_LEN * RANK * 4);
  float* to = (float*)take((size_t)S_LEN * RANK * 4);
  (void)ws_size; (void)in_sizes; (void)n_in; (void)out_size;

  const int nX = S_LEN * HDIM;         // 8.4M
  const int nWq = HDIM * HDIM;         // 16.8M
  const int nWk = N_KV * HD_ * HDIM;   // 4.2M

  // 1) casts
  cast_f32_bf16<<<2048, 256, 0, stream>>>(hidden, Xb, nX);
  cast_f32_bf16<<<2048, 256, 0, stream>>>(q_w, Wq, nWq);
  cast_f32_bf16<<<2048, 256, 0, stream>>>(k_w, Wk, nWk);
  cast_f32_bf16<<<2048, 256, 0, stream>>>(v_w, Wv, nWk);
  cast_f32_bf16<<<2048, 256, 0, stream>>>(o_w, Wo, nWq);

  // 2) LoRA t = X a^T
  lora_t_f32<<<S_LEN / 4, 64, 0, stream>>>(hidden, q_a, tq, HDIM);
  lora_t_f32<<<S_LEN / 4, 64, 0, stream>>>(hidden, k_a, tk, HDIM);
  lora_t_f32<<<S_LEN / 4, 64, 0, stream>>>(hidden, v_a, tv, HDIM);

  // 3) projections
  gemm_lora<1><<<dim3(HDIM / 64, S_LEN / 128), 256, 0, stream>>>(
      Xb, Wq, tq, q_b, nullptr, Qb, HDIM, HDIM, 0);
  gemm_lora<1><<<dim3(N_KV * HD_ / 64, S_LEN / 128), 256, 0, stream>>>(
      Xb, Wk, tk, k_b, nullptr, Kb, N_KV * HD_, HDIM, 0);
  gemm_lora<2><<<dim3(N_KV * HD_ / 64, S_LEN / 128), 256, 0, stream>>>(
      Xb, Wv, tv, v_b, nullptr, Vt, N_KV * HD_, HDIM, S_LEN);

  // 4) RoPE (in place)
  {
    int tQ = S_LEN * N_H * 64;
    int tK = S_LEN * N_KV * 64;
    rope_kernel<<<(tQ + 255) / 256, 256, 0, stream>>>(Qb, posids, N_H, HDIM);
    rope_kernel<<<(tK + 255) / 256, 256, 0, stream>>>(Kb, posids, N_KV, N_KV * HD_);
  }

  // 5) attention
  attn_kernel<<<dim3(S_LEN / 64, N_H), 128, 0, stream>>>(Qb, Kb, Vt, AO);

  // 6) output projection (+LoRA) -> fp32 d_out
  lora_t_bf16<<<S_LEN / 4, 64, 0, stream>>>(AO, o_a, to, HDIM);
  gemm_lora<0><<<dim3(HDIM / 64, S_LEN / 128), 256, 0, stream>>>(
      AO, Wo, to, o_b, outF, nullptr, HDIM, HDIM, 0);
}
